// MultiHeadSelfAttention_44994077393076
// MI455X (gfx1250) — matrix operand order repaired and checked by reading, not compiled, not run on hardware
//
#include <hip/hip_runtime.h>
#include <hip/hip_bf16.h>
#include <stdint.h>

// ---------------------------------------------------------------------------
// Problem constants (from reference)
// ---------------------------------------------------------------------------
static constexpr int BATCH = 16384;
static constexpr int DIN   = 2048;
static constexpr int DK    = 2048;
static constexpr int DV    = 2048;
static constexpr int NH    = 16;
static constexpr int DHEAD = 128;   // dk == dv == 128
static constexpr int DOUT  = 512;

typedef __attribute__((ext_vector_type(16))) __bf16 v16bf;
typedef __attribute__((ext_vector_type(8)))  float  v8f;

union FragU { v16bf v; uint4 u[2]; };

__device__ __forceinline__ unsigned short f2bf(float f) {
  unsigned u = __builtin_bit_cast(unsigned, f);
  u = (u + 0x7FFFu + ((u >> 16) & 1u)) >> 16;   // round-to-nearest-even
  return (unsigned short)u;
}
__device__ __forceinline__ float bf2f(unsigned short h) {
  unsigned u = ((unsigned)h) << 16;
  return __builtin_bit_cast(float, u);
}

// ---------------------------------------------------------------------------
// Optional CDNA5 async global->LDS path (ASYNCcnt-tracked, no VGPR staging)
// ---------------------------------------------------------------------------
#if defined(__has_builtin)
#if __has_builtin(__builtin_amdgcn_global_load_async_to_lds_b128) && \
    __has_builtin(__builtin_amdgcn_s_wait_asynccnt)
#define USE_ASYNC_LDS 1
#endif
#if __has_builtin(__builtin_amdgcn_sched_barrier)
#define SCHED_FENCE() __builtin_amdgcn_sched_barrier(0)
#endif
#endif
#ifndef SCHED_FENCE
#define SCHED_FENCE()
#endif

#ifdef USE_ASYNC_LDS
typedef int v4i_vec __attribute__((vector_size(16)));
__device__ __forceinline__ void async_b128(void* lds, const void* g) {
  __builtin_amdgcn_global_load_async_to_lds_b128(
      (__attribute__((address_space(1))) v4i_vec*)(g),
      (__attribute__((address_space(3))) v4i_vec*)(lds), 0, 0);
}
#endif

// ---------------------------------------------------------------------------
// Generic bf16 WMMA GEMM:  C[M,N] = epilogue( A[M,K] @ B[K,N] )
//   A   : bf16 row-major, leading dim lda, per-z-batch element offset aBatch
//   Bt  : bf16, B transposed => [N,K] row-major (ld == K), batch offset bBatch
//   Block: 256 thr = 8 waves; tile 128x128, K-step 32, double-buffered LDS.
//   Wave w computes rows [16w,16w+16) x 128 cols -> 8 wmma accumulators.
// Epilogues:
//   0: bf16 store            1: feat = relu(x)+1e-4, bf16
//   2: scale by z[row*16+z], bf16            3: +bias, store f32 AND bf16
//   4: g=sigmoid(x+bias); out=g*aux0+(1-g)*aux1, bf16     5: +bias, store f32
// ---------------------------------------------------------------------------
template <int EPI>
__global__ __launch_bounds__(256) void gemm_bf16_k(
    const unsigned short* __restrict__ A, long long aBatch, int lda,
    const unsigned short* __restrict__ Bt, long long bBatch,
    unsigned short* __restrict__ Cb, long long cBatch, int ldc,
    float* __restrict__ Cf, int ldcf,
    const float* __restrict__ bias,
    const float* __restrict__ zscale,
    const float* __restrict__ aux0,
    const float* __restrict__ aux1,
    int M, int N, int K)
{
  // +8 halfs pad: rows 80B apart -> 16B aligned, conflict-free b128 access
  __shared__ unsigned short As[2][128][40];
  __shared__ unsigned short Bs[2][128][40];

  const int tid  = threadIdx.x;
  const int lane = tid & 31;
  const int wave = tid >> 5;
  const int tileN = blockIdx.x * 128;
  const int tileM = blockIdx.y * 128;
  const int head  = blockIdx.z;

  A  += (long long)head * aBatch;
  Bt += (long long)head * bBatch;
  if (Cb) Cb += (long long)head * cBatch;

  v8f acc[8] = {};

  const int lr = tid >> 1;          // 0..127: tile row this thread loads
  const int lp = (tid & 1) * 16;    // half-row part (16 halfs = 32B)
  const unsigned short* gA = A  + (size_t)(tileM + lr) * lda + lp;
  const unsigned short* gB = Bt + (size_t)(tileN + lr) * K   + lp;

  const int am  = wave * 16 + (lane & 15);
  const int akb = (lane < 16) ? 0 : 8;
  const int bkh = (lane < 16) ? 0 : 8;

#ifdef USE_ASYNC_LDS
  // -------- double-buffered async pipeline --------
  async_b128(&As[0][lr][lp],     gA);
  async_b128(&As[0][lr][lp + 8], gA + 8);
  async_b128(&Bs[0][lr][lp],     gB);
  async_b128(&Bs[0][lr][lp + 8], gB + 8);
  int buf = 0;
  for (int k0 = 0; k0 < K; k0 += 32) {
    if (k0 + 32 < K) {
      const unsigned short* na = gA + k0 + 32;
      const unsigned short* nb = gB + k0 + 32;
      async_b128(&As[buf ^ 1][lr][lp],     na);
      async_b128(&As[buf ^ 1][lr][lp + 8], na + 8);
      async_b128(&Bs[buf ^ 1][lr][lp],     nb);
      async_b128(&Bs[buf ^ 1][lr][lp + 8], nb + 8);
      __builtin_amdgcn_s_wait_asynccnt(4);   // previous tile's 4 copies done
    } else {
      __builtin_amdgcn_s_wait_asynccnt(0);
    }
    __syncthreads();
    FragU fa;
    fa.u[0] = *(const uint4*)&As[buf][am][akb];
    fa.u[1] = *(const uint4*)&As[buf][am][akb + 16];
    FragU fb[8];
#pragma unroll
    for (int nt = 0; nt < 8; ++nt) {
      const int bn = nt * 16 + (lane & 15);
      fb[nt].u[0] = *(const uint4*)&Bs[buf][bn][bkh];
      fb[nt].u[1] = *(const uint4*)&Bs[buf][bn][bkh + 16];
    }
    SCHED_FENCE();   // keep all ds_loads above the wmma chain (one dscnt wait)
#pragma unroll
    for (int nt = 0; nt < 8; ++nt)
      acc[nt] = __builtin_amdgcn_wmma_f32_16x16x32_bf16(
          false, fa.v, false, fb[nt].v, (short)0, acc[nt], false, false);
    SCHED_FENCE();
    __syncthreads();
    buf ^= 1;
  }
#else
  // -------- fallback: sync staging through VGPRs --------
  for (int k0 = 0; k0 < K; k0 += 32) {
    const uint4* ga = (const uint4*)(gA + k0);
    const uint4* gb = (const uint4*)(gB + k0);
    uint4 a0 = ga[0], a1 = ga[1];
    uint4 b0 = gb[0], b1 = gb[1];
    __syncthreads();
    *(uint4*)&As[0][lr][lp]     = a0;  *(uint4*)&As[0][lr][lp + 8] = a1;
    *(uint4*)&Bs[0][lr][lp]     = b0;  *(uint4*)&Bs[0][lr][lp + 8] = b1;
    __syncthreads();
    FragU fa;
    fa.u[0] = *(const uint4*)&As[0][am][akb];
    fa.u[1] = *(const uint4*)&As[0][am][akb + 16];
    FragU fb[8];
#pragma unroll
    for (int nt = 0; nt < 8; ++nt) {
      const int bn = nt * 16 + (lane & 15);
      fb[nt].u[0] = *(const uint4*)&Bs[0][bn][bkh];
      fb[nt].u[1] = *(const uint4*)&Bs[0][bn][bkh + 16];
    }
    SCHED_FENCE();
#pragma unroll
    for (int nt = 0; nt < 8; ++nt)
      acc[nt] = __builtin_amdgcn_wmma_f32_16x16x32_bf16(
          false, fa.v, false, fb[nt].v, (short)0, acc[nt], false, false);
    SCHED_FENCE();
  }
#endif

  // C/D layout: VGPR r -> row r (lanes 0-15) / row r+8 (lanes 16-31), col = lane%16
  const int rowBase = tileM + wave * 16 + ((lane >> 4) << 3);
#pragma unroll
  for (int nt = 0; nt < 8; ++nt) {
    const int col = tileN + nt * 16 + (lane & 15);
#pragma unroll
    for (int r = 0; r < 8; ++r) {
      const int row = rowBase + r;
      float val = acc[nt][r];
      if constexpr (EPI == 0) {
        Cb[(size_t)row * ldc + col] = f2bf(val);
      } else if constexpr (EPI == 1) {
        val = fmaxf(val, 0.0f) + 1e-4f;
        Cb[(size_t)row * ldc + col] = f2bf(val);
      } else if constexpr (EPI == 2) {
        val *= zscale[(size_t)row * NH + head];
        Cb[(size_t)row * ldc + col] = f2bf(val);
      } else if constexpr (EPI == 3) {
        val += bias[col];
        Cf[(size_t)row * ldcf + col] = val;
        Cb[(size_t)row * ldc + col]  = f2bf(val);
      } else if constexpr (EPI == 4) {
        const float g  = 1.0f / (1.0f + __expf(-(val + bias[col])));
        const float op = aux0[(size_t)row * DIN + col];
        const float xx = aux1[(size_t)row * DIN + col];
        Cb[(size_t)row * ldc + col] = f2bf(g * op + (1.0f - g) * xx);
      } else {  // EPI == 5
        Cf[(size_t)row * ldcf + col] = val + bias[col];
      }
    }
  }
}

// ---------------------------------------------------------------------------
// kv[h] = sum_b k[b,h,:] (x) v[b,h,:]   (per-head 128x128, split-K over B,
// WMMA over staged-transposed tiles, fp32 atomic reduction into kvacc)
// grid: (BATCH/bChunk, NH)
// ---------------------------------------------------------------------------
__global__ __launch_bounds__(256) void kv_outer_k(
    const unsigned short* __restrict__ kb, const unsigned short* __restrict__ vb,
    float* __restrict__ kvacc, int bChunk)
{
  __shared__ unsigned short Kt[128][40];   // [d][b]  == (K_h)^T tile
  __shared__ unsigned short Vt[128][40];   // [e][b]  == V_h tile transposed

  const int tid  = threadIdx.x;
  const int lane = tid & 31;
  const int wave = tid >> 5;
  const int head = blockIdx.y;
  const int b0   = blockIdx.x * bChunk;

  v8f acc[8] = {};

  const int lb = tid >> 3;          // 0..31 local batch row
  const int d0 = (tid & 7) * 16;    // feature start (16 halfs)
  const int am  = wave * 16 + (lane & 15);
  const int akb = (lane < 16) ? 0 : 8;
  const int bkh = (lane < 16) ? 0 : 8;

  for (int bb = 0; bb < bChunk; bb += 32) {
    const size_t grow = (size_t)(b0 + bb + lb);
    const uint4* gk = (const uint4*)(kb + grow * DK + head * DHEAD + d0);
    const uint4* gv = (const uint4*)(vb + grow * DV + head * DHEAD + d0);
    uint4 k0v = gk[0], k1v = gk[1];
    uint4 v0v = gv[0], v1v = gv[1];
    unsigned short tk[16], tv[16];
    *(uint4*)&tk[0] = k0v;  *(uint4*)&tk[8] = k1v;
    *(uint4*)&tv[0] = v0v;  *(uint4*)&tv[8] = v1v;
    __syncthreads();
#pragma unroll
    for (int j = 0; j < 16; ++j) {
      Kt[d0 + j][lb] = tk[j];
      Vt[d0 + j][lb] = tv[j];
    }
    __syncthreads();

    FragU fa;
    fa.u[0] = *(const uint4*)&Kt[am][akb];
    fa.u[1] = *(const uint4*)&Kt[am][akb + 16];
    FragU fb[8];
#pragma unroll
    for (int nt = 0; nt < 8; ++nt) {
      const int bn = nt * 16 + (lane & 15);
      fb[nt].u[0] = *(const uint4*)&Vt[bn][bkh];
      fb[nt].u[1] = *(const uint4*)&Vt[bn][bkh + 16];
    }
    SCHED_FENCE();
#pragma unroll
    for (int nt = 0; nt < 8; ++nt)
      acc[nt] = __builtin_amdgcn_wmma_f32_16x16x32_bf16(
          false, fa.v, false, fb[nt].v, (short)0, acc[nt], false, false);
    SCHED_FENCE();
  }

  float* dst = kvacc + (size_t)head * DHEAD * DHEAD;
  const int rowBase = wave * 16 + ((lane >> 4) << 3);
#pragma unroll
  for (int nt = 0; nt < 8; ++nt) {
    const int col = nt * 16 + (lane & 15);
#pragma unroll
    for (int r = 0; r < 8; ++r)
      atomicAdd(&dst[(size_t)(rowBase + r) * DHEAD + col], acc[nt][r]);
  }
}

// ksum[c] = sum_b k[b,c]   (c = 0..2047, per-head contiguous)
__global__ __launch_bounds__(256) void ksum_k(
    const unsigned short* __restrict__ kb, float* __restrict__ ksum)
{
  const int r0 = blockIdx.x * 128;
  const int c  = threadIdx.x * 8;
  float s[8] = {};
  for (int rr = 0; rr < 128; ++rr) {
    const uint4 d = *(const uint4*)(kb + (size_t)(r0 + rr) * DK + c);
    const unsigned short* h = (const unsigned short*)&d;
#pragma unroll
    for (int j = 0; j < 8; ++j) s[j] += bf2f(h[j]);
  }
#pragma unroll
  for (int j = 0; j < 8; ++j) atomicAdd(&ksum[c + j], s[j]);
}

// z[b,h] = 1 / (dot(q[b,h,:], ksum[h,:]) + 1e-6)
__global__ void zcalc_k(const unsigned short* __restrict__ qb,
                        const float* __restrict__ ksum, float* __restrict__ z)
{
  const int idx = blockIdx.x * blockDim.x + threadIdx.x;  // b*NH + h
  const int b = idx >> 4, h = idx & 15;
  const unsigned short* qr = qb + (size_t)b * DK + h * DHEAD;
  const float* ks = ksum + h * DHEAD;
  float s = 0.0f;
  for (int d = 0; d < DHEAD; d += 8) {
    uint4 u = *(const uint4*)(qr + d);
    const unsigned short* hh = (const unsigned short*)&u;
#pragma unroll
    for (int j = 0; j < 8; ++j) s += bf2f(hh[j]) * ks[d + j];
  }
  z[idx] = 1.0f / (s + 1e-6f);
}

// fp32 [R,C] -> bf16 [C,R] (transpose + downconvert), batched via blockIdx.z
__global__ __launch_bounds__(256) void transp_k(
    const float* __restrict__ src, unsigned short* __restrict__ dst,
    int R, int C, long long sBatch, long long dBatch)
{
  __shared__ float tile[32][33];
  src += (long long)blockIdx.z * sBatch;
  dst += (long long)blockIdx.z * dBatch;
  const int cx = blockIdx.x * 32 + threadIdx.x;
#pragma unroll
  for (int i = 0; i < 4; ++i) {
    const int r = blockIdx.y * 32 + threadIdx.y + i * 8;
    tile[threadIdx.y + i * 8][threadIdx.x] = src[(size_t)r * C + cx];
  }
  __syncthreads();
#pragma unroll
  for (int i = 0; i < 4; ++i) {
    const int drow = blockIdx.x * 32 + threadIdx.y + i * 8;  // C index
    const int dcol = blockIdx.y * 32 + threadIdx.x;          // R index
    dst[(size_t)drow * R + dcol] = f2bf(tile[threadIdx.x][threadIdx.y + i * 8]);
  }
}

// x fp32 -> xb bf16, and also right half of concat buffer catb[:, 2048:4096]
__global__ void xconv_k(const float* __restrict__ x,
                        unsigned short* __restrict__ xb,
                        unsigned short* __restrict__ catb)
{
  const size_t i = (size_t)blockIdx.x * blockDim.x + threadIdx.x;
  const size_t b = i >> 11;          // / DIN
  const int    c = (int)(i & (DIN - 1));
  const unsigned short h = f2bf(x[i]);
  xb[i] = h;
  catb[b * (2 * DIN) + DIN + c] = h;
}

__global__ void zero_k(float* __restrict__ p, int n) {
  const int i = blockIdx.x * blockDim.x + threadIdx.x;
  if (i < n) p[i] = 0.0f;
}

// ---------------------------------------------------------------------------
extern "C" void kernel_launch(void* const* d_in, const int* in_sizes, int n_in,
                              void* d_out, int out_size, void* d_ws, size_t ws_size,
                              hipStream_t stream) {
  (void)in_sizes; (void)n_in; (void)out_size; (void)ws_size;
  const float* x    = (const float*)d_in[0];
  const float* Wq   = (const float*)d_in[1];
  const float* Wk   = (const float*)d_in[2];
  const float* Wv   = (const float*)d_in[3];
  const float* Wo   = (const float*)d_in[4];
  const float* bo   = (const float*)d_in[5];
  const float* Wg   = (const float*)d_in[6];
  const float* bg   = (const float*)d_in[7];
  const float* Wout = (const float*)d_in[8];
  const float* bout = (const float*)d_in[9];

  char* ws = (char*)d_ws;
  size_t off = 0;
  auto alloc = [&](size_t bytes) -> void* {
    off = (off + 255) & ~(size_t)255;
    void* p = ws + off;
    off += bytes;
    return p;
  };

  unsigned short* xb    = (unsigned short*)alloc((size_t)BATCH * DIN * 2);
  unsigned short* catb  = (unsigned short*)alloc((size_t)BATCH * 2 * DIN * 2);
  unsigned short* qb    = (unsigned short*)alloc((size_t)BATCH * DK * 2);
  unsigned short* kbuf  = (unsigned short*)alloc((size_t)BATCH * DK * 2);
  unsigned short* vbuf  = (unsigned short*)alloc((size_t)BATCH * DV * 2);
  unsigned short* attb  = (unsigned short*)alloc((size_t)BATCH * DV * 2);
  unsigned short* outfb = (unsigned short*)alloc((size_t)BATCH * DIN * 2);
  unsigned short* Wqt   = (unsigned short*)alloc((size_t)DIN * DK * 2);
  unsigned short* Wkt   = (unsigned short*)alloc((size_t)DIN * DK * 2);
  unsigned short* Wvt   = (unsigned short*)alloc((size_t)DIN * DV * 2);
  unsigned short* Wot   = (unsigned short*)alloc((size_t)DV * DIN * 2);
  unsigned short* Wgt   = (unsigned short*)alloc((size_t)2 * DIN * DIN * 2);
  unsigned short* Woutt = (unsigned short*)alloc((size_t)DIN * DOUT * 2);
  unsigned short* kvbt  = (unsigned short*)alloc((size_t)NH * DHEAD * DHEAD * 2);
  float* outproj = (float*)alloc((size_t)BATCH * DIN * 4);
  float* kvf     = (float*)alloc(((size_t)NH * DHEAD * DHEAD + DK) * 4);
  float* ksum    = kvf + (size_t)NH * DHEAD * DHEAD;   // contiguous for joint zeroing
  float* zbuf    = (float*)alloc((size_t)BATCH * NH * 4);

  const dim3 tb(32, 8);
  const int nz = NH * DHEAD * DHEAD + DK;
  zero_k<<<(nz + 255) / 256, 256, 0, stream>>>(kvf, nz);
  xconv_k<<<(BATCH * DIN) / 256, 256, 0, stream>>>(x, xb, catb);

  transp_k<<<dim3(DK / 32, DIN / 32, 1), tb, 0, stream>>>(Wq, Wqt, DIN, DK, 0, 0);
  transp_k<<<dim3(DK / 32, DIN / 32, 1), tb, 0, stream>>>(Wk, Wkt, DIN, DK, 0, 0);
  transp_k<<<dim3(DV / 32, DIN / 32, 1), tb, 0, stream>>>(Wv, Wvt, DIN, DV, 0, 0);
  transp_k<<<dim3(DIN / 32, DV / 32, 1), tb, 0, stream>>>(Wo, Wot, DV, DIN, 0, 0);
  transp_k<<<dim3(DIN / 32, (2 * DIN) / 32, 1), tb, 0, stream>>>(Wg, Wgt, 2 * DIN, DIN, 0, 0);
  transp_k<<<dim3(DOUT / 32, DIN / 32, 1), tb, 0, stream>>>(Wout, Woutt, DIN, DOUT, 0, 0);

  // q = feat(x@Wq), k = feat(x@Wk), v = x@Wv
  const dim3 gQKV(DK / 128, BATCH / 128, 1);
  gemm_bf16_k<1><<<gQKV, 256, 0, stream>>>(xb, 0, DIN, Wqt, 0, qb,   0, DK, nullptr, 0,
                                           nullptr, nullptr, nullptr, nullptr, BATCH, DK, DIN);
  gemm_bf16_k<1><<<gQKV, 256, 0, stream>>>(xb, 0, DIN, Wkt, 0, kbuf, 0, DK, nullptr, 0,
                                           nullptr, nullptr, nullptr, nullptr, BATCH, DK, DIN);
  gemm_bf16_k<0><<<gQKV, 256, 0, stream>>>(xb, 0, DIN, Wvt, 0, vbuf, 0, DV, nullptr, 0,
                                           nullptr, nullptr, nullptr, nullptr, BATCH, DV, DIN);

  // kv (split-K over batch, atomic f32 reduce), ksum, z
  kv_outer_k<<<dim3(8, NH, 1), 256, 0, stream>>>(kbuf, vbuf, kvf, BATCH / 8);
  ksum_k<<<BATCH / 128, 256, 0, stream>>>(kbuf, ksum);
  zcalc_k<<<(BATCH * NH) / 256, 256, 0, stream>>>(qb, ksum, zbuf);
  // kv fp32 [d,e] -> bf16 [e,d] per head (GEMM wants Bt = kv^T)
  transp_k<<<dim3(4, 4, NH), tb, 0, stream>>>(kvf, kvbt, DHEAD, DHEAD,
                                              DHEAD * DHEAD, DHEAD * DHEAD);

  // att = (q @ kv) * z    batched over heads (blockIdx.z)
  gemm_bf16_k<2><<<dim3(1, BATCH / 128, NH), 256, 0, stream>>>(
      qb, DHEAD, DK, kvbt, DHEAD * DHEAD, attb, DHEAD, DV, nullptr, 0,
      nullptr, zbuf, nullptr, nullptr, BATCH, DHEAD, DHEAD);

  // out_proj = att @ Wo + bo  (f32 + bf16 copy into left half of concat)
  gemm_bf16_k<3><<<dim3(DIN / 128, BATCH / 128, 1), 256, 0, stream>>>(
      attb, 0, DV, Wot, 0, catb, 0, 2 * DIN, outproj, DIN,
      bo, nullptr, nullptr, nullptr, BATCH, DIN, DV);

  // gate = sigmoid([out_proj|x] @ Wg + bg); out = gate*out_proj + (1-gate)*x
  gemm_bf16_k<4><<<dim3(DIN / 128, BATCH / 128, 1), 256, 0, stream>>>(
      catb, 0, 2 * DIN, Wgt, 0, outfb, 0, DIN, nullptr, 0,
      bg, nullptr, outproj, x, BATCH, DIN, 2 * DIN);

  // result = out @ Wout + bout  -> d_out (fp32)
  gemm_bf16_k<5><<<dim3(DOUT / 128, BATCH / 128, 1), 256, 0, stream>>>(
      outfb, 0, DIN, Woutt, 0, nullptr, 0, 0, (float*)d_out, DOUT,
      bout, nullptr, nullptr, nullptr, BATCH, DOUT, DIN);
}
